// ExpertParallelSwitchTransformerBase_2199023256084
// MI455X (gfx1250) — compile-verified, matrix operand order checked by reading, not run on hardware
//
#include <hip/hip_runtime.h>
#include <cstdint>

// ---------------- problem constants (match reference) ----------------
#define EXPERTS 8
#define DMODEL  1024
#define DFF     4096
#define TOK     8192          // BATCH*SEQ = 4*2048
#define CAP     1280          // int(1.25 * 8192 / 8)
#define SLOTS   (EXPERTS * CAP)

// ---------------- CDNA5 WMMA types ----------------
typedef __attribute__((ext_vector_type(16))) __bf16 v16bf;
typedef __attribute__((ext_vector_type(8)))  __bf16 bf16x8;
typedef __attribute__((ext_vector_type(8)))  float  v8f;

union FragBF { v16bf v; bf16x8 h[2]; };

// =====================================================================
// 1) slot_token init
// =====================================================================
__global__ void moe_init_slots(int* __restrict__ slot_token) {
  const int s = blockIdx.x * 256 + threadIdx.x;
  if (s < SLOTS) slot_token[s] = -1;
}

// =====================================================================
// 2) Router: one wave per token. logits = x_t . Wr, softmax max/argmax.
// =====================================================================
__global__ __launch_bounds__(256) void moe_router(
    const float* __restrict__ x, const float* __restrict__ Wr,
    int* __restrict__ eidx, float* __restrict__ pmax) {
  const int lane = threadIdx.x & 31;
  const int tok  = blockIdx.x * 8 + (threadIdx.x >> 5);
  const float* xr = x + (size_t)tok * DMODEL;

  float l[EXPERTS];
#pragma unroll
  for (int e = 0; e < EXPERTS; ++e) l[e] = 0.f;

  for (int d = lane; d < DMODEL; d += 32) {
    const float xv = xr[d];
    const float4 w0 = *(const float4*)&Wr[d * EXPERTS + 0];
    const float4 w1 = *(const float4*)&Wr[d * EXPERTS + 4];
    l[0] = __builtin_fmaf(xv, w0.x, l[0]);
    l[1] = __builtin_fmaf(xv, w0.y, l[1]);
    l[2] = __builtin_fmaf(xv, w0.z, l[2]);
    l[3] = __builtin_fmaf(xv, w0.w, l[3]);
    l[4] = __builtin_fmaf(xv, w1.x, l[4]);
    l[5] = __builtin_fmaf(xv, w1.y, l[5]);
    l[6] = __builtin_fmaf(xv, w1.z, l[6]);
    l[7] = __builtin_fmaf(xv, w1.w, l[7]);
  }
#pragma unroll
  for (int e = 0; e < EXPERTS; ++e) {
#pragma unroll
    for (int off = 16; off > 0; off >>= 1) l[e] += __shfl_xor(l[e], off, 32);
  }
  if (lane == 0) {
    float mx = l[0]; int am = 0;
#pragma unroll
    for (int e = 1; e < EXPERTS; ++e)
      if (l[e] > mx) { mx = l[e]; am = e; }
    float s = 0.f;
#pragma unroll
    for (int e = 0; e < EXPERTS; ++e) s += __expf(l[e] - mx);
    eidx[tok] = am;
    pmax[tok] = 1.f / s;   // exp(lmax-mx)/sum = 1/sum
  }
}

// =====================================================================
// 3) Deterministic first-come capacity assignment (single block).
// =====================================================================
__global__ __launch_bounds__(256) void moe_scan(
    const int* __restrict__ eidx, int* __restrict__ slot_token) {
  __shared__ int cnt[256][EXPERTS];
  const int tid = threadIdx.x;
  const int t0  = tid * (TOK / 256);   // 32 tokens each

  int local[EXPERTS];
#pragma unroll
  for (int e = 0; e < EXPERTS; ++e) local[e] = 0;
  for (int i = 0; i < TOK / 256; ++i) local[eidx[t0 + i]]++;
#pragma unroll
  for (int e = 0; e < EXPERTS; ++e) cnt[tid][e] = local[e];
  __syncthreads();

  if (tid < EXPERTS) {            // serial exclusive prefix per expert
    int run = 0;
    for (int j = 0; j < 256; ++j) { const int c = cnt[j][tid]; cnt[j][tid] = run; run += c; }
  }
  __syncthreads();

  int off[EXPERTS];
#pragma unroll
  for (int e = 0; e < EXPERTS; ++e) off[e] = cnt[tid][e];
  for (int i = 0; i < TOK / 256; ++i) {
    const int tok = t0 + i;
    const int e   = eidx[tok];
    const int pos = off[e]++;
    if (pos < CAP) slot_token[e * CAP + pos] = tok;   // kept tokens only
  }
}

// =====================================================================
// 4) Dispatch: Xd[slot][D] bf16 = x[token] or zeros (empty slot)
// =====================================================================
__global__ __launch_bounds__(256) void moe_dispatch(
    const float* __restrict__ x, const int* __restrict__ slot_token,
    __bf16* __restrict__ Xd) {
  const int s = blockIdx.x;
  const int t = slot_token[s];
  __bf16* dst = Xd + (size_t)s * DMODEL;
  if (t >= 0) {
    const float* src = x + (size_t)t * DMODEL;
#pragma unroll
    for (int i = 0; i < DMODEL / 256; ++i) {
      const int d = threadIdx.x + 256 * i;
      dst[d] = (__bf16)src[d];
    }
  } else {
#pragma unroll
    for (int i = 0; i < DMODEL / 256; ++i)
      dst[threadIdx.x + 256 * i] = (__bf16)0.f;
  }
}

// =====================================================================
// 5) out = x  (residual path; GEMM2 epilogue overwrites kept token rows)
// =====================================================================
__global__ __launch_bounds__(256) void moe_out_init(
    const float* __restrict__ x, float* __restrict__ out) {
  const size_t i = ((size_t)blockIdx.x * 256 + threadIdx.x) * 4;
  *(float4*)&out[i] = *(const float4*)&x[i];
}

// =====================================================================
// 6/7) Tiled bf16 WMMA GEMM, double-buffered LDS, async A staging.
//   Workgroup tile 128(M) x 256(N), BK=32. 8 waves, each 64x64 =
//   4x4 v_wmma_f32_16x16x32_bf16 tiles (16 WMMA : 16 ds_load_b128).
//   A (bf16) staged global->LDS via global_load_async_to_lds_b128
//   (ASYNCcnt path); B (f32 weights) converted to bf16 + transposed to
//   [N][K] through VGPRs.
//   FUSE_OUT=false: Hout = relu(acc) as bf16.
//   FUSE_OUT=true : out[token] = acc * pmax[token] via slot_token map.
// =====================================================================
template<int KTOT, int NTOT, bool FUSE_OUT>
__global__ __launch_bounds__(256) void moe_gemm(
    const __bf16* __restrict__ A, const float* __restrict__ Wfull,
    __bf16* __restrict__ Hout, const int* __restrict__ slot_token,
    const float* __restrict__ pmax, float* __restrict__ out) {
  constexpr int BM = 128, BN = 256, BK = 32;
  constexpr int MT = CAP / BM;          // 10
  constexpr int NT = NTOT / BN;         // 16 (gemm1) or 4 (gemm2)
  constexpr int KTILES = KTOT / BK;

  __shared__ __align__(16) __bf16 la[2][BM * BK];   // [M][K], 2 x 8 KB
  __shared__ __align__(16) __bf16 lb[2][BN * BK];   // [N][K], 2 x 16 KB

  const int bx  = blockIdx.x;
  const int e   = bx / (MT * NT);
  const int rem = bx % (MT * NT);
  const int mt  = rem / NT, nt = rem % NT;
  const int rowbase = e * CAP + mt * BM;
  const int colbase = nt * BN;
  const float* __restrict__ W = Wfull + (size_t)e * KTOT * NTOT;

  const int tid  = threadIdx.x;
  const int lane = tid & 31;
  const int wv   = tid >> 5;
  const int wm   = wv & 1;              // 2 waves along M (64 rows each)
  const int wn   = wv >> 1;             // 4 waves along N (64 cols each)
  const int kb   = (lane < 16) ? 0 : 8; // ISA 16-bit A/B lane K split
  const int ln   = lane & 15;

  v8f acc[4][4];
  const v8f vzero = {};
#pragma unroll
  for (int i = 0; i < 4; ++i)
#pragma unroll
    for (int j = 0; j < 4; ++j) acc[i][j] = vzero;

  // A-chunk geometry (two 16B chunks per thread per K-step)
  const int r0 = tid >> 2,         o0 = (tid & 3) * 8;
  const int r1 = (tid + 256) >> 2, o1 = (tid & 3) * 8;

  auto stage = [&](int kt, int buf) {
    const int k0 = kt * BK;
    // ---- A: pure bf16 copy, CDNA5 async global->LDS (ASYNCcnt) ----
    {
      const uint32_t lbase = (uint32_t)(uintptr_t)&la[buf][0];
      const uint32_t la0 = lbase + (uint32_t)(r0 * BK + o0) * 2u;
      const uint32_t la1 = lbase + (uint32_t)(r1 * BK + o1) * 2u;
      const uint64_t g0 = (uint64_t)(uintptr_t)&A[(size_t)(rowbase + r0) * KTOT + k0 + o0];
      const uint64_t g1 = (uint64_t)(uintptr_t)&A[(size_t)(rowbase + r1) * KTOT + k0 + o1];
      asm volatile("global_load_async_to_lds_b128 %0, %1, off"
                   :: "v"(la0), "v"(g0) : "memory");
      asm volatile("global_load_async_to_lds_b128 %0, %1, off"
                   :: "v"(la1), "v"(g1) : "memory");
    }
    // ---- B: f32 load, cvt bf16, transpose into [N][K] ----
#pragma unroll
    for (int i = 0; i < (BK * BN) / (256 * 4); ++i) {   // 8 float4 per thread
      const int g  = tid + 256 * i;       // 2048 groups of 4 floats
      const int kk = g >> 6;              // BN/4 = 64 groups per K row
      const int nn = (g & 63) * 4;
      const float4 w4 = *(const float4*)&W[(size_t)(k0 + kk) * NTOT + colbase + nn];
      lb[buf][(nn + 0) * BK + kk] = (__bf16)w4.x;
      lb[buf][(nn + 1) * BK + kk] = (__bf16)w4.y;
      lb[buf][(nn + 2) * BK + kk] = (__bf16)w4.z;
      lb[buf][(nn + 3) * BK + kk] = (__bf16)w4.w;
    }
  };

  stage(0, 0);
  for (int kt = 0; kt < KTILES; ++kt) {
    asm volatile("s_wait_asynccnt 0x0" ::: "memory");  // drain async A stage
    __syncthreads();
    const int cur = kt & 1;
    if (kt + 1 < KTILES) stage(kt + 1, cur ^ 1);       // overlap with WMMAs

    FragBF fa[4];
#pragma unroll
    for (int i = 0; i < 4; ++i) {
      const int m = wm * 64 + i * 16 + ln;
      fa[i].h[0] = *(const bf16x8*)&la[cur][m * BK + kb];
      fa[i].h[1] = *(const bf16x8*)&la[cur][m * BK + kb + 16];
    }
#pragma unroll
    for (int j = 0; j < 4; ++j) {
      FragBF fb;
      const int n = wn * 64 + j * 16 + ln;
      fb.h[0] = *(const bf16x8*)&lb[cur][n * BK + kb];
      fb.h[1] = *(const bf16x8*)&lb[cur][n * BK + kb + 16];
#pragma unroll
      for (int i = 0; i < 4; ++i)
        acc[i][j] = __builtin_amdgcn_wmma_f32_16x16x32_bf16(
            false, fa[i].v, false, fb.v, (short)0, acc[i][j], false, false);
    }
  }

  // epilogue. C/D layout: VGPR r, lane l -> M = r + 8*(l>=16), N = l&15
  const int mh = (lane >= 16) ? 8 : 0;
#pragma unroll
  for (int i = 0; i < 4; ++i)
#pragma unroll
    for (int j = 0; j < 4; ++j) {
      const int n = colbase + wn * 64 + j * 16 + ln;
#pragma unroll
      for (int r = 0; r < 8; ++r) {
        const int m = rowbase + wm * 64 + i * 16 + mh + r;
        if constexpr (FUSE_OUT) {
          const int t = slot_token[m];
          if (t >= 0) out[(size_t)t * DMODEL + n] = acc[i][j][r] * pmax[t];
        } else {
          float v = acc[i][j][r];
          v = v > 0.f ? v : 0.f;
          Hout[(size_t)m * NTOT + n] = (__bf16)v;
        }
      }
    }
}

// =====================================================================
// launcher
// =====================================================================
extern "C" void kernel_launch(void* const* d_in, const int* in_sizes, int n_in,
                              void* d_out, int out_size, void* d_ws, size_t ws_size,
                              hipStream_t stream) {
  (void)in_sizes; (void)n_in; (void)out_size; (void)ws_size;
  const float* x  = (const float*)d_in[0];   // [T, D]
  const float* Wr = (const float*)d_in[1];   // [D, E]
  const float* W1 = (const float*)d_in[2];   // [E, D, F]
  const float* W2 = (const float*)d_in[3];   // [E, F, D]
  float* out = (float*)d_out;                // [T, D]

  char* ws = (char*)d_ws;
  int*    slot_token = (int*)(ws + 0);                          // 40 KB
  int*    eidx       = (int*)(ws + (64 << 10));                 // 32 KB
  float*  pmax       = (float*)(ws + (128 << 10));              // 32 KB
  __bf16* Xd         = (__bf16*)(ws + (1 << 20));               // 21 MB
  __bf16* H          = (__bf16*)(ws + (1 << 20) + (32u << 20)); // 84 MB

  moe_init_slots<<<(SLOTS + 255) / 256, 256, 0, stream>>>(slot_token);
  moe_router<<<TOK / 8, 256, 0, stream>>>(x, Wr, eidx, pmax);
  moe_scan<<<1, 256, 0, stream>>>(eidx, slot_token);
  moe_dispatch<<<SLOTS, 256, 0, stream>>>(x, slot_token, Xd);
  moe_out_init<<<(TOK * DMODEL) / (256 * 4), 256, 0, stream>>>(x, out);

  moe_gemm<DMODEL, DFF, false>
      <<<EXPERTS * (CAP / 128) * (DFF / 256), 256, 0, stream>>>(
          Xd, W1, H, nullptr, nullptr, nullptr);
  moe_gemm<DFF, DMODEL, true>
      <<<EXPERTS * (CAP / 128) * (DMODEL / 256), 256, 0, stream>>>(
          H, W2, nullptr, slot_token, pmax, out);
}
